// BGALayer_4879082848623
// MI455X (gfx1250) — matrix operand order repaired and checked by reading, not compiled, not run on hardware
//
#include <hip/hip_runtime.h>

// ---------------------------------------------------------------------------
// BGA layer for MI455X (gfx1250): bf16 WMMA GEMMs + fused attention kernels.
// Shapes: N=102400 nodes, C=128, H=8 heads (dh=16), P=3200 patches, S=32.
// ---------------------------------------------------------------------------

constexpr int kN = 102400;
constexpr int kC = 128;
constexpr int kH = 8;
constexpr int kP = 3200;
constexpr int kS = 32;

typedef __attribute__((ext_vector_type(16))) __bf16 v16bf;
typedef __attribute__((ext_vector_type(8)))  float  v8f;

union Frag {
  v16bf v;
  uint4 q[2];
  unsigned short u[16];
};

__device__ __forceinline__ unsigned short f2bf(float f) {
  unsigned int x = __float_as_uint(f);
  x += 0x7FFFu + ((x >> 16) & 1u);           // round-to-nearest-even
  return (unsigned short)(x >> 16);
}

__device__ __forceinline__ v8f vzero8() {
  v8f z;
#pragma unroll
  for (int i = 0; i < 8; ++i) z[i] = 0.0f;
  return z;
}

__device__ __forceinline__ v8f wmma_bf16(const Frag& a, const Frag& b, v8f c) {
  return __builtin_amdgcn_wmma_f32_16x16x32_bf16(false, a.v, false, b.v,
                                                 (short)0, c, false, false);
}

// ------------------------------- fp32 -> bf16 ------------------------------
__global__ __launch_bounds__(256)
void k_conv(const float* __restrict__ s, unsigned short* __restrict__ d, int n) {
  int i = blockIdx.x * 256 + threadIdx.x;
  if (i < n) d[i] = f2bf(s[i]);
}

// ------------------------------- row LayerNorm -----------------------------
// one wave per row of 128; 8 rows per block
__global__ __launch_bounds__(256)
void k_ln(const float* __restrict__ X, const float* __restrict__ g,
          const float* __restrict__ b, float eps,
          float* __restrict__ outF, unsigned short* __restrict__ outB, int M) {
  const int wave = threadIdx.x >> 5, lane = threadIdx.x & 31;
  const long m = (long)blockIdx.x * 8 + wave;
  if (m >= M) return;
  const float* row = X + (size_t)m * kC;
  float x[4]; float s = 0.f;
#pragma unroll
  for (int i = 0; i < 4; ++i) { x[i] = row[lane + i * 32]; s += x[i]; }
  for (int msk = 16; msk; msk >>= 1) s += __shfl_xor(s, msk, 32);
  float mu = s * (1.0f / 128.0f);
  float vs = 0.f;
#pragma unroll
  for (int i = 0; i < 4; ++i) { float d = x[i] - mu; vs += d * d; }
  for (int msk = 16; msk; msk >>= 1) vs += __shfl_xor(vs, msk, 32);
  float rinv = rsqrtf(vs * (1.0f / 128.0f) + eps);
#pragma unroll
  for (int i = 0; i < 4; ++i) {
    int c = lane + i * 32;
    float y = (x[i] - mu) * rinv * g[c] + b[c];
    if (outF) outF[(size_t)m * kC + c] = y;
    if (outB) outB[(size_t)m * kC + c] = f2bf(y);
  }
}

// ------------------------------- gather rows -------------------------------
__global__ __launch_bounds__(256)
void k_gather(const float* __restrict__ LN, const int* __restrict__ idx,
              float* __restrict__ pxF, unsigned short* __restrict__ pxB) {
  const size_t i = (size_t)blockIdx.x * 256 + threadIdx.x;
  const int r = (int)(i >> 7);
  const int c = (int)(i & 127);
  float v = LN[(size_t)idx[r] * kC + c];
  pxF[i] = v;
  pxB[i] = f2bf(v);
}

// -------------------------- patch mean + LayerNorm -------------------------
__global__ __launch_bounds__(128)
void k_mean_ln(const float* __restrict__ PX, const float* __restrict__ g,
               const float* __restrict__ b,
               float* __restrict__ outF, unsigned short* __restrict__ outB) {
  __shared__ float part[4];
  const int c = threadIdx.x;
  const int p = blockIdx.x;
  const int wave = c >> 5, lane = c & 31;
  float s = 0.0f;
  for (int r = 0; r < kS; ++r) s += PX[((size_t)p * kS + r) * kC + c];
  float val = s * (1.0f / 32.0f);
  float t = val;
  for (int msk = 16; msk; msk >>= 1) t += __shfl_xor(t, msk, 32);
  if (lane == 0) part[wave] = t;
  __syncthreads();
  float mu = (part[0] + part[1] + part[2] + part[3]) * (1.0f / 128.0f);
  float d = val - mu;
  float t2 = d * d;
  for (int msk = 16; msk; msk >>= 1) t2 += __shfl_xor(t2, msk, 32);
  __syncthreads();
  if (lane == 0) part[wave] = t2;
  __syncthreads();
  float var = (part[0] + part[1] + part[2] + part[3]) * (1.0f / 128.0f);
  float y = d * rsqrtf(var + 1e-5f) * g[c] + b[c];
  outF[(size_t)p * kC + c] = y;
  outB[(size_t)p * kC + c] = f2bf(y);
}

// ---------------------------- generic WMMA GEMM ----------------------------
// out[M,128] = epilogue( alpha * A[M,128]bf16 @ B[128,128]bf16 )
// epilogue: + bias[col] + rbias[(row/rdiv)*128+col]; relu; + resid[row,col];
// stores fp32 and/or bf16; optional row remap via out_idx (scatter).
// M must be a multiple of 128; grid.x = M/128, block = 256 (8 waves).
__global__ __launch_bounds__(256)
void k_gemm(const unsigned short* __restrict__ A,
            const unsigned short* __restrict__ Bw,
            const float* __restrict__ bias,
            const float* __restrict__ rbias, int rdiv,
            const float* __restrict__ resid,
            float* __restrict__ outF,
            unsigned short* __restrict__ outB,
            const int* __restrict__ out_idx,
            int M, float alpha, int relu) {
  (void)M;
  __shared__ __align__(16) unsigned short Bt[128][136];  // B transposed [n][k]
  const int tid = threadIdx.x;
  for (int e = tid; e < 128 * 128; e += 256) {
    Bt[e & 127][e >> 7] = Bw[e];
  }
  __syncthreads();
  const int wave = tid >> 5;
  const int lane = tid & 31;
  const int hh   = lane >> 4;    // 0: lanes 0-15, 1: lanes 16-31
  const int l16  = lane & 15;
  const long m0  = (long)blockIdx.x * 128 + wave * 16;
  const unsigned short* Arow = A + (size_t)(m0 + l16) * kC;
  v8f acc[8];
#pragma unroll
  for (int j = 0; j < 8; ++j) acc[j] = vzero8();
#pragma unroll
  for (int kb = 0; kb < 128; kb += 32) {
    Frag af;  // A layout: low half K={kb..kb+7, kb+16..23}, high half +8
    af.q[0] = *(const uint4*)(Arow + kb + hh * 8);
    af.q[1] = *(const uint4*)(Arow + kb + hh * 8 + 16);
#pragma unroll
    for (int j = 0; j < 8; ++j) {
      Frag bf;  // B layout: lane col = l16, K = {kb+hh*8..+7, kb+16+hh*8..+7}
      const unsigned short* bp = &Bt[j * 16 + l16][kb + hh * 8];
      bf.q[0] = *(const uint4*)(bp);
      bf.q[1] = *(const uint4*)(bp + 16);
      acc[j] = wmma_bf16(af, bf, acc[j]);
    }
  }
#pragma unroll
  for (int j = 0; j < 8; ++j) {
    const int col = j * 16 + l16;
    const float bia = bias ? bias[col] : 0.0f;
#pragma unroll
    for (int vv = 0; vv < 8; ++vv) {
      const long m = m0 + vv + hh * 8;   // C layout: M = vgpr + half*8
      float t = acc[j][vv] * alpha + bia;
      if (rbias) t += rbias[(size_t)(m / rdiv) * kC + col];
      if (relu) t = fmaxf(t, 0.0f);
      if (resid) t += resid[(size_t)m * kC + col];
      const long mo = out_idx ? (long)out_idx[m] : m;
      if (outF) outF[(size_t)mo * kC + col] = t;
      if (outB) outB[(size_t)mo * kC + col] = f2bf(t);
    }
  }
}

// ---------------------- per-patch MHA (S=32, dh=16) ------------------------
// block = patch (grid P), wave = head. q already scaled by 1/sqrt(dh).
__global__ __launch_bounds__(256)
void k_attn1(const unsigned short* __restrict__ Q,
             const unsigned short* __restrict__ Kk,
             const unsigned short* __restrict__ V,
             unsigned short* __restrict__ O) {
  __shared__ __align__(16) unsigned short qs[32][136];
  __shared__ __align__(16) unsigned short ks[32][136];
  __shared__ __align__(16) unsigned short vs[32][136];
  __shared__ __align__(16) unsigned short ps[8][32][40];
  const int tid = threadIdx.x;
  const size_t base = (size_t)blockIdx.x * kS * kC;
  for (int e = tid * 8; e < kS * kC; e += 256 * 8) {
    int s = e >> 7, c = e & 127;
    *(uint4*)&qs[s][c] = *(const uint4*)&Q[base + e];
    *(uint4*)&ks[s][c] = *(const uint4*)&Kk[base + e];
    *(uint4*)&vs[s][c] = *(const uint4*)&V[base + e];
  }
  __syncthreads();
  const int wave = tid >> 5, lane = tid & 31;
  const int hh = lane >> 4, l16 = lane & 15;
  const int hc = wave * 16;  // head channel offset
  uint4 z4 = make_uint4(0u, 0u, 0u, 0u);

  // A frags of q (dh=16 padded to K=32 with zeros)
  Frag qa[2];
#pragma unroll
  for (int mi = 0; mi < 2; ++mi) {
    const unsigned short* qp = &qs[mi * 16 + l16][hc + hh * 8];
    qa[mi].q[0] = *(const uint4*)qp;
    qa[mi].q[1] = z4;  // K=16..31 pad
  }
  // B frags of k^T (real K=hh*8..hh*8+7 in elements 0-7; elements 8-15 are pad)
  Frag kf[2];
#pragma unroll
  for (int nj = 0; nj < 2; ++nj) {
    const unsigned short* kp = &ks[nj * 16 + l16][hc + hh * 8];
    kf[nj].q[0] = *(const uint4*)kp;
    kf[nj].q[1] = z4;
  }
  v8f st[2][2];
#pragma unroll
  for (int mi = 0; mi < 2; ++mi)
#pragma unroll
    for (int nj = 0; nj < 2; ++nj)
      st[mi][nj] = wmma_bf16(qa[mi], kf[nj], vzero8());

  // softmax rows (reduce over the 16-lane half = the 16 cols of a tile)
#pragma unroll
  for (int mi = 0; mi < 2; ++mi) {
#pragma unroll
    for (int vv = 0; vv < 8; ++vv) {
      float a = st[mi][0][vv], b = st[mi][1][vv];
      float r = fmaxf(a, b);
#pragma unroll
      for (int msk = 1; msk < 16; msk <<= 1) r = fmaxf(r, __shfl_xor(r, msk, 32));
      float e0 = __expf(a - r), e1 = __expf(b - r);
      float s = e0 + e1;
#pragma unroll
      for (int msk = 1; msk < 16; msk <<= 1) s += __shfl_xor(s, msk, 32);
      float inv = 1.0f / s;
      int row = mi * 16 + vv + hh * 8;
      ps[wave][row][l16]      = f2bf(e0 * inv);
      ps[wave][row][16 + l16] = f2bf(e1 * inv);
    }
  }
  // V as B frag: col = dh lane, K = key (32 real keys)
  Frag vf;
#pragma unroll
  for (int i = 0; i < 16; ++i) vf.u[i] = vs[hh * 8 + i + (i & 8)][hc + l16];
#pragma unroll
  for (int mi = 0; mi < 2; ++mi) {
    Frag pa;  // probs as A frag via LDS (layout conversion)
    const unsigned short* pp = &ps[wave][mi * 16 + l16][hh * 8];
    pa.q[0] = *(const uint4*)pp;
    pa.q[1] = *(const uint4*)(pp + 16);
    v8f ot = wmma_bf16(pa, vf, vzero8());
#pragma unroll
    for (int vv = 0; vv < 8; ++vv) {
      size_t row = (size_t)blockIdx.x * kS + mi * 16 + vv + hh * 8;
      O[row * kC + hc + l16] = f2bf(ot[vv]);
    }
  }
}

// -------------------- cross-patch flash attention (P=3200) -----------------
// grid = P/16 query blocks; wave = head; K/V blocks cooperatively staged.
__global__ __launch_bounds__(256)
void k_attn2(const unsigned short* __restrict__ Q,
             const unsigned short* __restrict__ Kk,
             const unsigned short* __restrict__ V,
             unsigned short* __restrict__ O) {
  __shared__ __align__(16) unsigned short kbuf[32][136];
  __shared__ __align__(16) unsigned short vbuf[32][136];
  __shared__ __align__(16) unsigned short ps[8][16][40];
  const int tid = threadIdx.x;
  const int wave = tid >> 5, lane = tid & 31;
  const int hh = lane >> 4, l16 = lane & 15;
  const int hc = wave * 16;
  const int q0 = blockIdx.x * 16;
  uint4 z4 = make_uint4(0u, 0u, 0u, 0u);
  Frag qa;
  qa.q[0] = *(const uint4*)&Q[(size_t)(q0 + l16) * kC + hc + hh * 8];
  qa.q[1] = z4;  // dh pad
  v8f o = vzero8();
  float mrow[8], lrow[8];
#pragma unroll
  for (int vv = 0; vv < 8; ++vv) { mrow[vv] = -1e30f; lrow[vv] = 0.0f; }

  for (int kb = 0; kb < kP; kb += 32) {
    __syncthreads();
    for (int e = tid * 8; e < 32 * kC; e += 2048) {
      int s = e >> 7, c = e & 127;
      *(uint4*)&kbuf[s][c] = *(const uint4*)&Kk[(size_t)(kb + s) * kC + c];
      *(uint4*)&vbuf[s][c] = *(const uint4*)&V[(size_t)(kb + s) * kC + c];
    }
    __syncthreads();
    v8f st[2];
#pragma unroll
    for (int nj = 0; nj < 2; ++nj) {
      Frag kf;
      const unsigned short* kp = &kbuf[nj * 16 + l16][hc + hh * 8];
      kf.q[0] = *(const uint4*)kp;
      kf.q[1] = z4;
      st[nj] = wmma_bf16(qa, kf, vzero8());
    }
#pragma unroll
    for (int vv = 0; vv < 8; ++vv) {
      float a = st[0][vv], b = st[1][vv];
      float r = fmaxf(a, b);
#pragma unroll
      for (int msk = 1; msk < 16; msk <<= 1) r = fmaxf(r, __shfl_xor(r, msk, 32));
      float mn = fmaxf(mrow[vv], r);
      float e0 = __expf(a - mn), e1 = __expf(b - mn);
      float s = e0 + e1;
#pragma unroll
      for (int msk = 1; msk < 16; msk <<= 1) s += __shfl_xor(s, msk, 32);
      float sc = __expf(mrow[vv] - mn);
      lrow[vv] = lrow[vv] * sc + s;
      mrow[vv] = mn;
      o[vv] *= sc;
      int row = vv + hh * 8;
      ps[wave][row][l16]      = f2bf(e0);
      ps[wave][row][16 + l16] = f2bf(e1);
    }
    Frag pa;
    const unsigned short* pp = &ps[wave][l16][hh * 8];
    pa.q[0] = *(const uint4*)pp;
    pa.q[1] = *(const uint4*)(pp + 16);
    Frag vf;
#pragma unroll
    for (int i = 0; i < 16; ++i) vf.u[i] = vbuf[hh * 8 + i + (i & 8)][hc + l16];
    o = wmma_bf16(pa, vf, o);  // accumulate P@V into C operand
  }
#pragma unroll
  for (int vv = 0; vv < 8; ++vv) {
    size_t row = (size_t)q0 + vv + hh * 8;
    O[row * kC + hc + l16] = f2bf(o[vv] / lrow[vv]);
  }
}

// ---------------------------------------------------------------------------
extern "C" void kernel_launch(void* const* d_in, const int* in_sizes, int n_in,
                              void* d_out, int out_size, void* d_ws, size_t ws_size,
                              hipStream_t stream) {
  (void)in_sizes; (void)n_in; (void)out_size; (void)ws_size;
  const float* x     = (const float*)d_in[0];
  const int*   patch = (const int*)  d_in[1];
  const float* nn_g  = (const float*)d_in[2];
  const float* nn_b  = (const float*)d_in[3];
  const float* wq1   = (const float*)d_in[4];
  const float* wk1   = (const float*)d_in[5];
  const float* wv1   = (const float*)d_in[6];
  const float* wo1   = (const float*)d_in[7];
  const float* f1w1  = (const float*)d_in[8];
  const float* f1b1  = (const float*)d_in[9];
  const float* f1w2  = (const float*)d_in[10];
  const float* f1b2  = (const float*)d_in[11];
  const float* f1g   = (const float*)d_in[12];
  const float* f1b   = (const float*)d_in[13];
  const float* png   = (const float*)d_in[14];
  const float* pnb   = (const float*)d_in[15];
  const float* wq2   = (const float*)d_in[16];
  const float* wk2   = (const float*)d_in[17];
  const float* wv2   = (const float*)d_in[18];
  const float* wo2   = (const float*)d_in[19];
  const float* f2w1  = (const float*)d_in[20];
  const float* f2b1  = (const float*)d_in[21];
  const float* f2w2  = (const float*)d_in[22];
  const float* f2b2  = (const float*)d_in[23];
  const float* f2g   = (const float*)d_in[24];
  const float* f2b   = (const float*)d_in[25];
  const float* fw    = (const float*)d_in[26];
  const float* fb    = (const float*)d_in[27];

  const size_t NC = (size_t)kN * kC;   // 13,107,200
  const size_t PC = (size_t)kP * kC;   // 409,600
  const int    CC2 = kC * kC;          // 16,384

  char* wsb = (char*)d_ws;
  size_t off = 0;
  auto alloc = [&](size_t bytes) -> void* {
    void* p = wsb + off;
    off = (off + bytes + 255) & ~(size_t)255;
    return p;
  };
  float* buf_px = (float*)alloc(NC * 4);   // current px (patch order, fp32)
  float* buf_t  = (float*)alloc(NC * 4);   // intermediate px (fp32)
  unsigned short* u0 = (unsigned short*)alloc(NC * 2);
  unsigned short* u1 = (unsigned short*)alloc(NC * 2);
  unsigned short* u2 = (unsigned short*)alloc(NC * 2);
  unsigned short* u3 = (unsigned short*)alloc(NC * 2);
  float* sp0 = (float*)alloc(PC * 4);
  float* sp1 = (float*)alloc(PC * 4);
  float* sp2 = (float*)alloc(PC * 4);
  float* spt = (float*)alloc(PC * 4);
  unsigned short* sbf0  = (unsigned short*)alloc(PC * 2);
  unsigned short* sqh   = (unsigned short*)alloc(PC * 2);
  unsigned short* skh   = (unsigned short*)alloc(PC * 2);
  unsigned short* svh   = (unsigned short*)alloc(PC * 2);
  unsigned short* sat   = (unsigned short*)alloc(PC * 2);
  unsigned short* sbf1  = (unsigned short*)alloc(PC * 2);
  unsigned short* sbf2  = (unsigned short*)alloc(PC * 2);
  unsigned short* sbfp2 = (unsigned short*)alloc(PC * 2);
  unsigned short* wb[13];
  const float* wsrc[13] = {wq1, wk1, wv1, wo1, f1w1, f1w2,
                           wq2, wk2, wv2, wo2, f2w1, f2w2, fw};
  for (int i = 0; i < 12; ++i) wb[i] = (unsigned short*)alloc((size_t)CC2 * 2);
  wb[12] = (unsigned short*)alloc((size_t)2 * CC2 * 2);   // fuse_w [256,128]

  const dim3 blk(256);
  // weights -> bf16
  for (int i = 0; i < 13; ++i) {
    int n = (i == 12) ? 2 * CC2 : CC2;
    k_conv<<<dim3((n + 255) / 256), blk, 0, stream>>>(wsrc[i], wb[i], n);
  }
  float* outF = (float*)d_out;
  // 1) node_norm: LN(x) -> d_out (base for scatter), then gather to patch order
  k_ln<<<dim3(kN / 8), blk, 0, stream>>>(x, nn_g, nn_b, 1e-5f, outF, nullptr, kN);
  k_gather<<<dim3((unsigned)(NC / 256)), blk, 0, stream>>>(outF, patch, buf_px, u0);
  // 2) per-patch MHA: q (pre-scaled by 1/sqrt(16)), k, v
  k_gemm<<<dim3(kN / 128), blk, 0, stream>>>(u0, wb[0], nullptr, nullptr, 1, nullptr,
                                             nullptr, u1, nullptr, kN, 0.25f, 0);
  k_gemm<<<dim3(kN / 128), blk, 0, stream>>>(u0, wb[1], nullptr, nullptr, 1, nullptr,
                                             nullptr, u2, nullptr, kN, 1.0f, 0);
  k_gemm<<<dim3(kN / 128), blk, 0, stream>>>(u0, wb[2], nullptr, nullptr, 1, nullptr,
                                             nullptr, u3, nullptr, kN, 1.0f, 0);
  k_attn1<<<dim3(kP), blk, 0, stream>>>(u1, u2, u3, u0);
  k_gemm<<<dim3(kN / 128), blk, 0, stream>>>(u0, wb[3], nullptr, nullptr, 1, buf_px,
                                             buf_t, nullptr, nullptr, kN, 1.0f, 0);
  // 3) node FFN
  k_ln<<<dim3(kN / 8), blk, 0, stream>>>(buf_t, f1g, f1b, 1e-6f, nullptr, u1, kN);
  k_gemm<<<dim3(kN / 128), blk, 0, stream>>>(u1, wb[4], f1b1, nullptr, 1, nullptr,
                                             nullptr, u2, nullptr, kN, 1.0f, 1);
  k_gemm<<<dim3(kN / 128), blk, 0, stream>>>(u2, wb[5], f1b2, nullptr, 1, buf_t,
                                             buf_px, nullptr, nullptr, kN, 1.0f, 0);
  // 4) patch mean + LN
  k_mean_ln<<<dim3(kP), dim3(128), 0, stream>>>(buf_px, png, pnb, sp0, sbf0);
  // 5) cross-patch MHA
  k_gemm<<<dim3(kP / 128), blk, 0, stream>>>(sbf0, wb[6], nullptr, nullptr, 1, nullptr,
                                             nullptr, sqh, nullptr, kP, 0.25f, 0);
  k_gemm<<<dim3(kP / 128), blk, 0, stream>>>(sbf0, wb[7], nullptr, nullptr, 1, nullptr,
                                             nullptr, skh, nullptr, kP, 1.0f, 0);
  k_gemm<<<dim3(kP / 128), blk, 0, stream>>>(sbf0, wb[8], nullptr, nullptr, 1, nullptr,
                                             nullptr, svh, nullptr, kP, 1.0f, 0);
  k_attn2<<<dim3(kP / 16), blk, 0, stream>>>(sqh, skh, svh, sat);
  k_gemm<<<dim3(kP / 128), blk, 0, stream>>>(sat, wb[9], nullptr, nullptr, 1, sp0,
                                             sp1, nullptr, nullptr, kP, 1.0f, 0);
  // 6) patch FFN
  k_ln<<<dim3(kP / 8), blk, 0, stream>>>(sp1, f2g, f2b, 1e-6f, nullptr, sbf1, kP);
  k_gemm<<<dim3(kP / 128), blk, 0, stream>>>(sbf1, wb[10], f2b1, nullptr, 1, nullptr,
                                             nullptr, sbf2, nullptr, kP, 1.0f, 1);
  k_gemm<<<dim3(kP / 128), blk, 0, stream>>>(sbf2, wb[11], f2b2, nullptr, 1, sp1,
                                             sp2, nullptr, nullptr, kP, 1.0f, 0);
  // 7) fuse: relu(px@Wtop + (p@Wbot + fuse_b)) + px, scatter rows by patch idx
  k_conv<<<dim3((unsigned)((PC + 255) / 256)), blk, 0, stream>>>(sp2, sbfp2, (int)PC);
  k_gemm<<<dim3(kP / 128), blk, 0, stream>>>(sbfp2, wb[12] + CC2, fb, nullptr, 1,
                                             nullptr, spt, nullptr, nullptr, kP, 1.0f, 0);
  k_conv<<<dim3((unsigned)((NC + 255) / 256)), blk, 0, stream>>>(buf_px, u1, (int)NC);
  k_gemm<<<dim3(kN / 128), blk, 0, stream>>>(u1, wb[12], nullptr, spt, kS, buf_px,
                                             outF, nullptr, patch, kN, 1.0f, 1);
}
